// TemporalStateGCN_71382356459942
// MI455X (gfx1250) — matrix operand order repaired and checked by reading, not compiled, not run on hardware
//
#include <hip/hip_runtime.h>
#include <hip/hip_bf16.h>

typedef __attribute__((ext_vector_type(16))) _Float16 v16h;
typedef __attribute__((ext_vector_type(8)))  _Float16 v8h;
typedef __attribute__((ext_vector_type(8)))  float    v8f;

__device__ __forceinline__ float wave_sum(float v) {
#pragma unroll
    for (int off = 16; off > 0; off >>= 1) v += __shfl_xor(v, off, 32);
    return v;
}

// ---------------------------------------------------------------------------
// f32 -> f16 conversion (grid-stride)
// ---------------------------------------------------------------------------
__global__ void cvt_f32_f16(const float* __restrict__ s, _Float16* __restrict__ d, size_t n) {
    size_t i = (size_t)blockIdx.x * blockDim.x + threadIdx.x;
    size_t stride = (size_t)gridDim.x * blockDim.x;
    for (; i < n; i += stride) d[i] = (_Float16)s[i];
}

// ---------------------------------------------------------------------------
// Generic C[M,Nc](f32) = A[M,K](f16,row-major,lda) @ W[Nc,K](f16,row-major,ldw)^T
// One wave per 16x16 output tile, 4 waves / block. K % 32 == 0, M % 16 == 0.
// ---------------------------------------------------------------------------
__global__ __launch_bounds__(128)
void gemm_f16_wmma(const _Float16* __restrict__ A, int lda,
                   const _Float16* __restrict__ W, int ldw,
                   float* __restrict__ C, int ldc,
                   int M, int Nc, int K)
{
    const int wid   = threadIdx.x >> 5;
    const int lane  = threadIdx.x & 31;
    const int jtile = blockIdx.y * 4 + wid;
    if (jtile * 16 >= Nc) return;
    const int rowBase = blockIdx.x * 16;
    const int m    = lane & 15;
    const int half = lane >> 4;

    const _Float16* arow = A + (size_t)(rowBase + m) * lda;
    const _Float16* wrow = W + (size_t)(jtile * 16 + m) * ldw;

    v8f acc = {};
    for (int k0 = 0; k0 < K; k0 += 32) {
        // A fragment: lane m holds K in [16*half, 16*half+16)
        v16h a = *(const v16h*)(arow + k0 + half * 16);
        // B fragment: lane holds column n = m, K in [16*half, 16*half+16),
        // contiguous along the weight row (C = A * W^T).
        v16h b = *(const v16h*)(wrow + k0 + half * 16);
        acc = __builtin_amdgcn_wmma_f32_16x16x32_f16(
                  false, a, false, b, (short)0, acc, false, false);
    }
    // D layout: VGPR i -> row (8*half + i), col m
    float* crow = C + (size_t)(rowBase + half * 8) * ldc + jtile * 16 + m;
#pragma unroll
    for (int i = 0; i < 8; ++i) crow[(size_t)i * ldc] = acc[i];
}

// ---------------------------------------------------------------------------
// Encoder epilogue: LN(relu) over 128 cols of epre, plus the 32-dim time
// encoder; writes h (f32 [N,160]) and hm16[:, :160] (f16, ld=320).
// One wave per node, 8 waves/block.
// ---------------------------------------------------------------------------
__global__ __launch_bounds__(256)
void encode_finalize(const float* __restrict__ epre,
                     const float* __restrict__ b_enc, const float* __restrict__ g_enc,
                     const float* __restrict__ be_enc,
                     const float* __restrict__ tsteps,
                     const float* __restrict__ w_time, const float* __restrict__ b_time,
                     const float* __restrict__ g_time, const float* __restrict__ be_time,
                     float* __restrict__ h, _Float16* __restrict__ hm16, int N)
{
    int node = blockIdx.x * 8 + (threadIdx.x >> 5);
    int lane = threadIdx.x & 31;
    if (node >= N) return;

    float v[4]; float s = 0.f, s2 = 0.f;
#pragma unroll
    for (int t = 0; t < 4; ++t) {
        int j = lane + 32 * t;
        float x = epre[(size_t)node * 128 + j] + b_enc[j];
        v[t] = x; s += x; s2 += x * x;
    }
    s = wave_sum(s); s2 = wave_sum(s2);
    float mean = s * (1.f / 128.f);
    float var  = s2 * (1.f / 128.f) - mean * mean;
    float rstd = rsqrtf(var + 1e-5f);
#pragma unroll
    for (int t = 0; t < 4; ++t) {
        int j = lane + 32 * t;
        float y = fmaxf((v[t] - mean) * rstd * g_enc[j] + be_enc[j], 0.f);
        h[(size_t)node * 160 + j] = y;
        hm16[(size_t)node * 320 + j] = (_Float16)y;
    }
    // time encoder: 32 dims, one per lane
    float tp = tsteps[node] * w_time[lane] + b_time[lane];
    float ts = wave_sum(tp), ts2 = wave_sum(tp * tp);
    float tm = ts * (1.f / 32.f);
    float tv = ts2 * (1.f / 32.f) - tm * tm;
    float tr = rsqrtf(tv + 1e-5f);
    float ty = fmaxf((tp - tm) * tr * g_time[lane] + be_time[lane], 0.f);
    h[(size_t)node * 160 + 128 + lane] = ty;
    hm16[(size_t)node * 320 + 128 + lane] = (_Float16)ty;
}

// ---------------------------------------------------------------------------
// In-degree counts (same for every layer)
// ---------------------------------------------------------------------------
__global__ void count_edges(const int* __restrict__ EI, float* __restrict__ cnt, int E) {
    int e = blockIdx.x * blockDim.x + threadIdx.x;
    if (e < E) atomicAdd(&cnt[EI[E + e]], 1.f);
}

// ---------------------------------------------------------------------------
// Per-edge message: pre = Apre[src] + Bpre[dst] + bm; LN; relu; *w; scatter-add.
// One wave per edge (5 cols/lane), shfl reductions for LN stats.
// ---------------------------------------------------------------------------
__global__ __launch_bounds__(256)
void edge_messages(const float* __restrict__ Apre, const float* __restrict__ Bpre,
                   const int* __restrict__ EI, const float* __restrict__ EW,
                   const float* __restrict__ bm, const float* __restrict__ gm,
                   const float* __restrict__ bem, float* __restrict__ msum, int E)
{
    int e = blockIdx.x * 8 + (threadIdx.x >> 5);
    int lane = threadIdx.x & 31;
    if (e >= E) return;
    int src = EI[e], dst = EI[E + e];
    float w = EW[e];
    const float* ar = Apre + (size_t)src * 160;
    const float* br = Bpre + (size_t)dst * 160;
    float v[5]; float s = 0.f, s2 = 0.f;
#pragma unroll
    for (int t = 0; t < 5; ++t) {
        int j = lane + 32 * t;
        float x = ar[j] + br[j] + bm[j];
        v[t] = x; s += x; s2 += x * x;
    }
    s = wave_sum(s); s2 = wave_sum(s2);
    float mean = s * (1.f / 160.f);
    float var  = s2 * (1.f / 160.f) - mean * mean;
    float rstd = rsqrtf(var + 1e-5f);
    float* mr = msum + (size_t)dst * 160;
#pragma unroll
    for (int t = 0; t < 5; ++t) {
        int j = lane + 32 * t;
        float y = fmaxf((v[t] - mean) * rstd * gm[j] + bem[j], 0.f) * w;
        atomicAdd(&mr[j], y);
    }
}

// ---------------------------------------------------------------------------
// messages = msum / (cnt+1e-8) * (cnt>0); write f16 into hm16[:, 160:320]
// ---------------------------------------------------------------------------
__global__ void msg_finalize(const float* __restrict__ msum, const float* __restrict__ cnt,
                             _Float16* __restrict__ hm16, int N)
{
    size_t idx = (size_t)blockIdx.x * blockDim.x + threadIdx.x;
    size_t total = (size_t)N * 160;
    if (idx >= total) return;
    size_t n = idx / 160;
    int j = (int)(idx - n * 160);
    float c = cnt[n];
    float m = (c > 0.f) ? msum[idx] / (c + 1e-8f) : 0.f;
    hm16[n * 320 + 160 + j] = (_Float16)m;
}

// ---------------------------------------------------------------------------
// Gated update: h_new = relu(LN(Upre+bu)); tw = sigmoid(h.wg + bg);
// h = tw*h_new + (1-tw)*h ; also refresh hm16[:, :160]. One wave per node.
// ---------------------------------------------------------------------------
__global__ __launch_bounds__(256)
void update_nodes(const float* __restrict__ Upre, const float* __restrict__ bu,
                  const float* __restrict__ gu, const float* __restrict__ beu,
                  const float* __restrict__ wg, const float* __restrict__ bg,
                  float* __restrict__ h, _Float16* __restrict__ hm16, int N)
{
    int node = blockIdx.x * 8 + (threadIdx.x >> 5);
    int lane = threadIdx.x & 31;
    if (node >= N) return;
    float* hr = h + (size_t)node * 160;
    const float* ur = Upre + (size_t)node * 160;
    float hv[5], u[5]; float s = 0.f, s2 = 0.f, dot = 0.f;
#pragma unroll
    for (int t = 0; t < 5; ++t) {
        int j = lane + 32 * t;
        float x = ur[j] + bu[j];
        u[t] = x; s += x; s2 += x * x;
        hv[t] = hr[j];
        dot += hv[t] * wg[j];
    }
    s = wave_sum(s); s2 = wave_sum(s2); dot = wave_sum(dot);
    float mean = s * (1.f / 160.f);
    float var  = s2 * (1.f / 160.f) - mean * mean;
    float rstd = rsqrtf(var + 1e-5f);
    float tw = 1.f / (1.f + __expf(-(dot + bg[0])));
    _Float16* hm = hm16 + (size_t)node * 320;
#pragma unroll
    for (int t = 0; t < 5; ++t) {
        int j = lane + 32 * t;
        float hn = fmaxf((u[t] - mean) * rstd * gu[j] + beu[j], 0.f);
        float o = tw * hn + (1.f - tw) * hv[t];
        hr[j] = o;
        hm[j] = (_Float16)o;
    }
}

// ---------------------------------------------------------------------------
// Output: add bias, L2-normalize rows of [N,128]. One wave per node.
// ---------------------------------------------------------------------------
__global__ __launch_bounds__(256)
void out_finalize(const float* __restrict__ Opre, const float* __restrict__ bo,
                  float* __restrict__ out, int N)
{
    int node = blockIdx.x * 8 + (threadIdx.x >> 5);
    int lane = threadIdx.x & 31;
    if (node >= N) return;
    const float* orow = Opre + (size_t)node * 128;
    float v[4]; float s2 = 0.f;
#pragma unroll
    for (int t = 0; t < 4; ++t) {
        int j = lane + 32 * t;
        float x = orow[j] + bo[j];
        v[t] = x; s2 += x * x;
    }
    s2 = wave_sum(s2);
    float inv = 1.f / fmaxf(sqrtf(s2), 1e-12f);
    float* od = out + (size_t)node * 128;
#pragma unroll
    for (int t = 0; t < 4; ++t) od[lane + 32 * t] = v[t] * inv;
}

// ---------------------------------------------------------------------------
extern "C" void kernel_launch(void* const* d_in, const int* in_sizes, int n_in,
                              void* d_out, int out_size, void* d_ws, size_t ws_size,
                              hipStream_t stream)
{
    (void)n_in; (void)out_size; (void)ws_size;
    const float* X    = (const float*)d_in[0];
    const int*   EI   = (const int*)  d_in[1];
    const float* EW   = (const float*)d_in[2];
    const float* TS   = (const float*)d_in[3];
    const float* Wenc = (const float*)d_in[4];
    const float* Benc = (const float*)d_in[5];
    const float* Genc = (const float*)d_in[6];
    const float* BEenc= (const float*)d_in[7];
    const float* Wt   = (const float*)d_in[8];
    const float* Bt   = (const float*)d_in[9];
    const float* Gt   = (const float*)d_in[10];
    const float* BEt  = (const float*)d_in[11];
    const float* WM   = (const float*)d_in[12];
    const float* BM   = (const float*)d_in[13];
    const float* GM   = (const float*)d_in[14];
    const float* BEM  = (const float*)d_in[15];
    const float* WU   = (const float*)d_in[16];
    const float* BU   = (const float*)d_in[17];
    const float* GU   = (const float*)d_in[18];
    const float* BEU  = (const float*)d_in[19];
    const float* WG   = (const float*)d_in[20];
    const float* BG   = (const float*)d_in[21];
    const float* WO   = (const float*)d_in[22];
    const float* BO   = (const float*)d_in[23];

    const int N = in_sizes[0] / 128;   // 100000
    const int E = in_sizes[2];         // 800000

    // --- workspace carving (256B aligned) ---
    char* p = (char*)d_ws;
    auto carve = [&](size_t bytes) -> void* {
        void* q = (void*)p;
        p += (bytes + 255) & ~(size_t)255;
        return q;
    };
    _Float16* Xf     = (_Float16*)carve((size_t)N * 128 * 2);
    _Float16* Wenc16 = (_Float16*)carve((size_t)128 * 128 * 2);
    _Float16* WM16   = (_Float16*)carve((size_t)2 * 160 * 320 * 2);
    _Float16* WU16   = (_Float16*)carve((size_t)2 * 160 * 320 * 2);
    _Float16* WO16   = (_Float16*)carve((size_t)128 * 160 * 2);
    float*    gbuf   = (float*)   carve((size_t)N * 160 * 4); // gemm outputs (epre/Upre/Opre)
    float*    h      = (float*)   carve((size_t)N * 160 * 4);
    _Float16* hm16   = (_Float16*)carve((size_t)N * 320 * 2); // [h_f16 | msg_f16]
    float*    Apre   = (float*)   carve((size_t)N * 160 * 4);
    float*    Bpre   = (float*)   carve((size_t)N * 160 * 4);
    float*    msum   = (float*)   carve((size_t)N * 160 * 4);
    float*    cnt    = (float*)   carve((size_t)N * 4);

    auto cvt = [&](const float* s, _Float16* d, size_t n) {
        int blocks = (int)((n + 1023) / 1024);
        if (blocks > 4096) blocks = 4096;
        cvt_f32_f16<<<blocks, 256, 0, stream>>>(s, d, n);
    };
    cvt(X, Xf, (size_t)N * 128);
    cvt(Wenc, Wenc16, (size_t)128 * 128);
    cvt(WM, WM16, (size_t)2 * 160 * 320);
    cvt(WU, WU16, (size_t)2 * 160 * 320);
    cvt(WO, WO16, (size_t)128 * 160);

    hipMemsetAsync(cnt, 0, (size_t)N * 4, stream);
    count_edges<<<(E + 255) / 256, 256, 0, stream>>>(EI, cnt, E);

    // encoder: epre = Xf @ Wenc^T  -> LN/relu + time encoder
    dim3 gEnc((N + 15) / 16, 2);
    gemm_f16_wmma<<<gEnc, 128, 0, stream>>>(Xf, 128, Wenc16, 128, gbuf, 128, N, 128, 128);
    encode_finalize<<<(N + 7) / 8, 256, 0, stream>>>(gbuf, Benc, Genc, BEenc,
                                                     TS, Wt, Bt, Gt, BEt, h, hm16, N);

    dim3 gD((N + 15) / 16, 3); // 160 cols -> 10 tiles over 3 blocks of 4 waves
    for (int l = 0; l < 2; ++l) {
        const _Float16* wm_l = WM16 + (size_t)l * 160 * 320;
        const _Float16* wu_l = WU16 + (size_t)l * 160 * 320;
        // message-MLP factorization: Apre = h@wm_src^T, Bpre = h@wm_dst^T
        gemm_f16_wmma<<<gD, 128, 0, stream>>>(hm16, 320, wm_l,       320, Apre, 160, N, 160, 160);
        gemm_f16_wmma<<<gD, 128, 0, stream>>>(hm16, 320, wm_l + 160, 320, Bpre, 160, N, 160, 160);
        hipMemsetAsync(msum, 0, (size_t)N * 160 * 4, stream);
        edge_messages<<<(E + 7) / 8, 256, 0, stream>>>(Apre, Bpre, EI, EW,
                                                       BM + l * 160, GM + l * 160, BEM + l * 160,
                                                       msum, E);
        msg_finalize<<<(unsigned)(((size_t)N * 160 + 255) / 256), 256, 0, stream>>>(msum, cnt, hm16, N);
        // update GEMM over concatenated [h|msg] (K = 320)
        gemm_f16_wmma<<<gD, 128, 0, stream>>>(hm16, 320, wu_l, 320, gbuf, 160, N, 160, 320);
        update_nodes<<<(N + 7) / 8, 256, 0, stream>>>(gbuf, BU + l * 160, GU + l * 160, BEU + l * 160,
                                                      WG + l * 160, BG + l, h, hm16, N);
    }

    // output projection + L2 normalize
    dim3 gOut((N + 15) / 16, 2);
    gemm_f16_wmma<<<gOut, 128, 0, stream>>>(hm16, 320, WO16, 160, gbuf, 128, N, 128, 160);
    out_finalize<<<(N + 7) / 8, 256, 0, stream>>>(gbuf, BO, (float*)d_out, N);
}
